// CausalSelfAttention_74156905333200
// MI455X (gfx1250) — matrix operand order repaired and checked by reading, not compiled, not run on hardware
//
#include <hip/hip_runtime.h>
#include <hip/hip_bf16.h>

// ---------------- types for WMMA ----------------
typedef __attribute__((ext_vector_type(16))) __bf16        v16bf;
typedef __attribute__((ext_vector_type(8)))  float         v8f;
typedef __attribute__((ext_vector_type(8)))  unsigned short u16x8;

union V16 {
    v16bf b;
    u16x8 h[2];
};

__device__ __forceinline__ v8f wmma_bf16(const V16& a, const V16& b, v8f c) {
    // D = A(16x32 bf16) * B(32x16 bf16) + C(16x16 f32)
    return __builtin_amdgcn_wmma_f32_16x16x32_bf16(
        /*neg_a=*/false, a.b, /*neg_b=*/false, b.b,
        /*c_mod=*/(short)0, c, /*reuse_a=*/false, /*reuse_b=*/false);
}

__device__ __forceinline__ unsigned short f2bf(float f) {
    union { float f; unsigned int u; } c; c.f = f;
    unsigned int r = c.u + 0x7fffu + ((c.u >> 16) & 1u);   // round-to-nearest-even
    return (unsigned short)(r >> 16);
}

// ---------------- problem constants ----------------
#define BATCH 2
#define SEQ   4096
#define EMB   768
#define E3    2304
#define NH    12
#define HD    64
#define MROWS (BATCH * SEQ)      // 8192

// workspace layout (bytes)
#define OFF_XB     0u                     // bf16 x   [8192][768]   (reused as ybf later)
#define OFF_WQKVT  12582912u              // bf16 WqkvT [2304][768]
#define OFF_WPROJT 16121856u              // bf16 WprojT [768][768]
#define OFF_QB     17301504u              // bf16 Q  [B][H][T][64]
#define OFF_KB     29884416u              // bf16 K  [B][H][T][64]
#define OFF_VT     42467328u              // bf16 Vt [B][H][64][T]
// total = 55050240 bytes (~52.5 MB)

// ---------------- convert kernels ----------------
__global__ void k_cvt(const float* __restrict__ src, unsigned short* __restrict__ dst, int n) {
    int i = blockIdx.x * blockDim.x + threadIdx.x;
    if (i < n) dst[i] = f2bf(src[i]);
}

// src [K][N] row-major f32 -> dst [N][K] row-major bf16
__global__ void k_cvt_t(const float* __restrict__ src, unsigned short* __restrict__ dst,
                        int K, int N) {
    int i = blockIdx.x * blockDim.x + threadIdx.x;
    if (i < K * N) {
        int k = i / N, n = i - k * N;
        dst[n * K + k] = f2bf(src[i]);
    }
}

// ---------------- QKV GEMM:  qkv = x @ Wqkv + b ----------------
// grid (64, 36), block 256 (8 waves). Wave w computes rows [blk.x*128+w*16, +16), cols [blk.y*64, +64)
__global__ __launch_bounds__(256)
void k_qkv(const unsigned short* __restrict__ xb,    // [8192][768] bf16
           const unsigned short* __restrict__ wT,    // [2304][768] bf16
           const float* __restrict__ bias,           // [2304] f32
           unsigned short* __restrict__ Qb,
           unsigned short* __restrict__ Kb,
           unsigned short* __restrict__ Vt) {
    const int lane = threadIdx.x & 31;
    const int wave = threadIdx.x >> 5;
    const int hf = lane >> 4, ln = lane & 15;
    const int mbase = blockIdx.x * 128 + wave * 16;
    const int nbase = blockIdx.y * 64;

    v8f acc[4] = {};
    for (int kk = 0; kk < EMB; kk += 32) {
        V16 a;
        const unsigned short* ap = xb + (size_t)(mbase + ln) * EMB + kk;
        a.h[0] = *(const u16x8*)(ap + hf * 16);        // K = kk + hf*16 .. +7
        a.h[1] = *(const u16x8*)(ap + hf * 16 + 8);    // K = kk+hf*16+8 .. +7
#pragma unroll
        for (int t = 0; t < 4; ++t) {
            V16 bm;
            const unsigned short* bp = wT + (size_t)(nbase + t * 16 + ln) * EMB + kk + hf * 16;
            bm.h[0] = *(const u16x8*)bp;
            bm.h[1] = *(const u16x8*)(bp + 8);
            acc[t] = wmma_bf16(a, bm, acc[t]);
        }
    }
#pragma unroll
    for (int t = 0; t < 4; ++t) {
        int c = nbase + t * 16 + ln;                   // output column (fixed per lane)
        float bv = bias[c];
#pragma unroll
        for (int r = 0; r < 8; ++r) {
            int m = mbase + r + 8 * hf;                // global row
            unsigned short bf = f2bf(acc[t][r] + bv);
            int b = m >> 12, tt = m & (SEQ - 1);
            if (c < EMB) {
                int h = c / HD, d = c % HD;
                Qb[(((size_t)(b * NH + h) * SEQ) + tt) * HD + d] = bf;
            } else if (c < 2 * EMB) {
                int c2 = c - EMB, h = c2 / HD, d = c2 % HD;
                Kb[(((size_t)(b * NH + h) * SEQ) + tt) * HD + d] = bf;
            } else {
                int c2 = c - 2 * EMB, h = c2 / HD, d = c2 % HD;
                Vt[(((size_t)(b * NH + h) * HD) + d) * SEQ + tt] = bf;   // V transposed
            }
        }
    }
}

// ---------------- flash attention: one wave per 16-query tile ----------------
// grid 768, block 256 (8 independent waves, NO workgroup barriers)
__global__ __launch_bounds__(256)
void k_attn(const unsigned short* __restrict__ Qb,
            const unsigned short* __restrict__ Kb,
            const unsigned short* __restrict__ Vt,
            unsigned short* __restrict__ ybf) {    // [8192][768] bf16
    __shared__ __align__(16) unsigned short plds[8][16][32];  // per-wave P staging

    const int lane = threadIdx.x & 31;
    const int wave = threadIdx.x >> 5;
    const int hf = lane >> 4, ln = lane & 15;

    const int tile = blockIdx.x * 8 + wave;          // [0, 6144)
    const int qt = tile & (SEQ / 16 - 1);
    const int bh = tile / (SEQ / 16);                // b*12+h
    const int qbase = qt * 16;

    const unsigned short* q = Qb + (size_t)bh * SEQ * HD;
    const unsigned short* k = Kb + (size_t)bh * SEQ * HD;
    const unsigned short* v = Vt + (size_t)bh * HD * SEQ;

    // Q A-operands: two k-steps (d 0..31, 32..63)
    V16 aq[2];
#pragma unroll
    for (int s = 0; s < 2; ++s) {
        const unsigned short* qp = q + (size_t)(qbase + ln) * HD + s * 32;
        aq[s].h[0] = *(const u16x8*)(qp + hf * 16);
        aq[s].h[1] = *(const u16x8*)(qp + hf * 16 + 8);
    }

    float mrun[8], lrun[8];
#pragma unroll
    for (int r = 0; r < 8; ++r) { mrun[r] = -3.0e38f; lrun[r] = 0.0f; }
    v8f o[4] = {};
    const float scale = 0.125f;                      // 1/sqrt(64)

    for (int kb = 0; kb < qbase + 16; kb += 32) {
        // ---- S = Q K^T for 32 keys (two 16-key N tiles) ----
        v8f s0 = {}, s1 = {};
#pragma unroll
        for (int s = 0; s < 2; ++s) {
            V16 b0, b1;
            const unsigned short* kp0 = k + (size_t)(kb + ln) * HD + s * 32 + hf * 16;
            const unsigned short* kp1 = k + (size_t)(kb + 16 + ln) * HD + s * 32 + hf * 16;
            b0.h[0] = *(const u16x8*)kp0; b0.h[1] = *(const u16x8*)(kp0 + 8);
            b1.h[0] = *(const u16x8*)kp1; b1.h[1] = *(const u16x8*)(kp1 + 8);
            s0 = wmma_bf16(aq[s], b0, s0);
            s1 = wmma_bf16(aq[s], b1, s1);
        }
        // ---- online softmax (row m = r + 8*hf lives across the 16 lanes of this half) ----
#pragma unroll
        for (int r = 0; r < 8; ++r) {
            int m = qbase + r + 8 * hf;
            float x0 = (kb + ln <= m)      ? s0[r] * scale : -3.0e38f;
            float x1 = (kb + 16 + ln <= m) ? s1[r] * scale : -3.0e38f;
            float rm = fmaxf(x0, x1);
#pragma unroll
            for (int off = 8; off >= 1; off >>= 1)
                rm = fmaxf(rm, __shfl_xor(rm, off, 32));
            float mnew  = fmaxf(mrun[r], rm);
            float alpha = __expf(mrun[r] - mnew);
            float p0 = __expf(x0 - mnew);
            float p1 = __expf(x1 - mnew);
            float rs = p0 + p1;
#pragma unroll
            for (int off = 8; off >= 1; off >>= 1)
                rs += __shfl_xor(rs, off, 32);
            lrun[r] = lrun[r] * alpha + rs;
            mrun[r] = mnew;
#pragma unroll
            for (int dt = 0; dt < 4; ++dt) o[dt][r] *= alpha;
            int row = r + 8 * hf;
            plds[wave][row][ln]      = f2bf(p0);
            plds[wave][row][ln + 16] = f2bf(p1);
        }
        asm volatile("s_wait_dscnt 0" ::: "memory");   // same-wave LDS RAW fence
        // ---- P as A-operand (16 x 32keys) ----
        V16 pa;
        pa.h[0] = *(const u16x8*)&plds[wave][ln][hf * 16];
        pa.h[1] = *(const u16x8*)&plds[wave][ln][hf * 16 + 8];
        // ---- O += P V  (4 d-subtiles of 16) ----
#pragma unroll
        for (int dt = 0; dt < 4; ++dt) {
            V16 vb;
            const unsigned short* vp = v + (size_t)(dt * 16 + ln) * SEQ + kb + hf * 16;
            vb.h[0] = *(const u16x8*)vp; vb.h[1] = *(const u16x8*)(vp + 8);
            o[dt] = wmma_bf16(pa, vb, o[dt]);
        }
    }

    // ---- epilogue: O /= l, write ybf[B*T][768] ----
    int h = bh % NH;
    int rowbase = (bh / NH) * SEQ + qbase;
#pragma unroll
    for (int r = 0; r < 8; ++r) {
        float inv = 1.0f / lrun[r];
        size_t row = (size_t)(rowbase + r + 8 * hf);
#pragma unroll
        for (int dt = 0; dt < 4; ++dt)
            ybf[row * EMB + h * HD + dt * 16 + ln] = f2bf(o[dt][r] * inv);
    }
}

// ---------------- output projection: out = y @ Wproj + b ----------------
// grid (64, 12), block 256
__global__ __launch_bounds__(256)
void k_proj(const unsigned short* __restrict__ ybf,   // [8192][768] bf16
            const unsigned short* __restrict__ wT,    // [768][768] bf16 (transposed)
            const float* __restrict__ bias,           // [768]
            float* __restrict__ out) {                // [8192][768] f32
    const int lane = threadIdx.x & 31;
    const int wave = threadIdx.x >> 5;
    const int hf = lane >> 4, ln = lane & 15;
    const int mbase = blockIdx.x * 128 + wave * 16;
    const int nbase = blockIdx.y * 64;

    v8f acc[4] = {};
    for (int kk = 0; kk < EMB; kk += 32) {
        V16 a;
        const unsigned short* ap = ybf + (size_t)(mbase + ln) * EMB + kk;
        a.h[0] = *(const u16x8*)(ap + hf * 16);
        a.h[1] = *(const u16x8*)(ap + hf * 16 + 8);
#pragma unroll
        for (int t = 0; t < 4; ++t) {
            V16 bm;
            const unsigned short* bp = wT + (size_t)(nbase + t * 16 + ln) * EMB + kk + hf * 16;
            bm.h[0] = *(const u16x8*)bp;
            bm.h[1] = *(const u16x8*)(bp + 8);
            acc[t] = wmma_bf16(a, bm, acc[t]);
        }
    }
#pragma unroll
    for (int t = 0; t < 4; ++t) {
        int c = nbase + t * 16 + ln;
        float bv = bias[c];
#pragma unroll
        for (int r = 0; r < 8; ++r) {
            size_t m = (size_t)(mbase + r + 8 * hf);
            out[m * EMB + c] = acc[t][r] + bv;
        }
    }
}

// ---------------- launch ----------------
extern "C" void kernel_launch(void* const* d_in, const int* in_sizes, int n_in,
                              void* d_out, int out_size, void* d_ws, size_t ws_size,
                              hipStream_t stream) {
    const float* x     = (const float*)d_in[0];   // [2,4096,768]
    const float* Wqkv  = (const float*)d_in[1];   // [768,2304]
    const float* bqkv  = (const float*)d_in[2];   // [2304]
    const float* Wproj = (const float*)d_in[3];   // [768,768]
    const float* bproj = (const float*)d_in[4];   // [768]
    float* out = (float*)d_out;

    char* ws = (char*)d_ws;                       // needs ~55 MB
    unsigned short* xb     = (unsigned short*)(ws + OFF_XB);
    unsigned short* wqkvT  = (unsigned short*)(ws + OFF_WQKVT);
    unsigned short* wprojT = (unsigned short*)(ws + OFF_WPROJT);
    unsigned short* Qb     = (unsigned short*)(ws + OFF_QB);
    unsigned short* Kb     = (unsigned short*)(ws + OFF_KB);
    unsigned short* Vt     = (unsigned short*)(ws + OFF_VT);
    unsigned short* ybf    = (unsigned short*)(ws + OFF_XB);  // reuse xb (dead after QKV)

    int nx = MROWS * EMB;
    k_cvt  <<<(nx + 255) / 256, 256, 0, stream>>>(x, xb, nx);
    k_cvt_t<<<(EMB * E3 + 255) / 256, 256, 0, stream>>>(Wqkv, wqkvT, EMB, E3);
    k_cvt_t<<<(EMB * EMB + 255) / 256, 256, 0, stream>>>(Wproj, wprojT, EMB, EMB);

    k_qkv <<<dim3(MROWS / 128, E3 / 64), 256, 0, stream>>>(xb, wqkvT, bqkv, Qb, Kb, Vt);
    k_attn<<<(BATCH * NH * SEQ / 16) / 8, 256, 0, stream>>>(Qb, Kb, Vt, ybf);
    k_proj<<<dim3(MROWS / 128, EMB / 64), 256, 0, stream>>>(ybf, wprojT, bproj, out);
}
